// Lattice_44315472560399
// MI455X (gfx1250) — compile-verified, matrix operand order checked by reading, not run on hardware
//
#include <hip/hip_runtime.h>

// ---- CDNA5 (gfx1250) wave32 WMMA types ----
typedef __bf16 bf16_t;
typedef __bf16 v16bf __attribute__((ext_vector_type(16)));
typedef float  v8f   __attribute__((ext_vector_type(8)));

static constexpr int IN_FEATURES  = 16;
static constexpr int OUT_FEATURES = 64;
static constexpr int BATCH        = 4096;
static constexpr int NBASIS       = 1 << IN_FEATURES;    // 65536
static constexpr int HPAD         = 260;                 // fHigh row stride (f32)
static constexpr int LPAD         = 264;                 // fLow row stride (bf16)
static constexpr int MTILE        = 16;                  // rows per workgroup (one WMMA tile)

// ---------------------------------------------------------------------------
// Kernel 1: one-time f32 -> bf16 conversion of W (row-major [64][65536]).
// ---------------------------------------------------------------------------
__global__ void wcvt_kernel(const float* __restrict__ W,
                            bf16_t* __restrict__ Wb, int n) {
  int i = (blockIdx.x * blockDim.x + threadIdx.x) * 4;
  if (i + 3 < n) {
    float4 v = *(const float4*)(W + i);
    Wb[i + 0] = (bf16_t)v.x;
    Wb[i + 1] = (bf16_t)v.y;
    Wb[i + 2] = (bf16_t)v.z;
    Wb[i + 3] = (bf16_t)v.w;
  }
}

// ---------------------------------------------------------------------------
// Kernel 2: fused feature-construction + GEMM.
//   out[n,o] = sum_h fHigh[n,h] * ( sum_l fLow[n,l] * W[o, h*256+l] ) + bias[o]
// One wave = one 16x16 output tile. The 8 fLow A-fragments (h-invariant) are
// loaded from LDS once and pinned in VGPRs; W B-fragments stream per h.
// ---------------------------------------------------------------------------
template <bool PRECONV>
__global__ void lattice_gemm(const float* __restrict__ x,
                             const void*  __restrict__ Wsrc,  // bf16 if PRECONV else f32
                             const float* __restrict__ bias,
                             float* __restrict__ out) {
  extern __shared__ char smem_raw[];
  float*  sHigh = (float*)smem_raw;                               // [16][HPAD] f32
  bf16_t* sLowB = (bf16_t*)(smem_raw + MTILE * HPAD * 4);         // [16][LPAD] bf16

  const int tid = threadIdx.x;           // 0..127 (4 waves)
  const int g   = blockIdx.x;            // M-tile index (16 rows)

  // ---- Phase 1: per-row half-products into LDS (8 threads per row) ------
  {
    const int r    = tid >> 3;           // row in tile, 0..15
    const int sub  = tid & 7;
    const int hsel = sub >> 2;           // 0: x[0..7] -> fHigh(f32); 1: x[8..15] -> fLow(bf16)
    const int jq   = sub & 3;            // quarter of the 256 basis values
    const float* xp = x + (size_t)(g * MTILE + r) * IN_FEATURES + hsel * 8;
    float4 va = *(const float4*)(xp);
    float4 vb = *(const float4*)(xp + 4);
    float xv[8] = {va.x, va.y, va.z, va.w, vb.x, vb.y, vb.z, vb.w};
    float om[8];
#pragma unroll
    for (int i = 0; i < 8; ++i) om[i] = 1.0f - xv[i];
    const int j0 = jq * 64;
    for (int j = j0; j < j0 + 64; ++j) {
      float p = 1.0f;
#pragma unroll
      for (int i = 0; i < 8; ++i)        // bit (7-i) of j selects factor i
        p *= ((j >> (7 - i)) & 1) ? xv[i] : om[i];
      if (hsel) sLowB[r * LPAD + j] = (bf16_t)p;
      else      sHigh[r * HPAD + j] = p;
    }
  }
  __syncthreads();

  // ---- Phase 2: WMMA main loop ------------------------------------------
  const int lane = tid & 31;
  const int wv   = tid >> 5;             // wave id = N-tile, 0..3
  const int hs   = lane >> 4;            // lane half-select (K grouping)
  const int nl   = lane & 15;
  const int o    = wv * 16 + nl;         // this lane's output column

  union BF { uint4 q[2]; v16bf v; };

  // fLow A-fragments: h-invariant, loaded once, resident in 64 VGPRs.
  BF afr[8];
  {
    const bf16_t* lp = sLowB + nl * LPAD + hs * 8;
#pragma unroll
    for (int kbl = 0; kbl < 8; ++kbl) {
      afr[kbl].q[0] = *(const uint4*)(lp + kbl * 32);        // K 0..7  (+16..23)
      afr[kbl].q[1] = *(const uint4*)(lp + kbl * 32 + 16);
    }
  }

  v8f acc = {};
  const float* fhBase = sHigh + (size_t)(hs * 8) * HPAD;     // rows hs*8 + r

#pragma unroll 1
  for (int h = 0; h < 256; ++h) {
    // B fragments for this h: W[o, h*256 .. h*256+255]
    BF bfr[8];
    if (PRECONV) {
      const bf16_t* bp = (const bf16_t*)Wsrc + (size_t)o * NBASIS + h * 256 + hs * 8;
#pragma unroll
      for (int kbl = 0; kbl < 8; ++kbl) {
        bfr[kbl].q[0] = *(const uint4*)(bp + kbl * 32);
        bfr[kbl].q[1] = *(const uint4*)(bp + kbl * 32 + 16);
      }
      __builtin_prefetch((const void*)(bp + 256), 0, 0);     // next h's chunk
    } else {
      const float* bp = (const float*)Wsrc + (size_t)o * NBASIS + h * 256 + hs * 8;
#pragma unroll
      for (int kbl = 0; kbl < 8; ++kbl) {
        float4 c0 = *(const float4*)(bp + kbl * 32);
        float4 c1 = *(const float4*)(bp + kbl * 32 + 4);
        float4 c2 = *(const float4*)(bp + kbl * 32 + 16);
        float4 c3 = *(const float4*)(bp + kbl * 32 + 20);
        float bc[16] = {c0.x, c0.y, c0.z, c0.w, c1.x, c1.y, c1.z, c1.w,
                        c2.x, c2.y, c2.z, c2.w, c3.x, c3.y, c3.z, c3.w};
#pragma unroll
        for (int e = 0; e < 16; ++e) bfr[kbl].v[e] = (bf16_t)bc[e];
      }
    }

    // Two interleaved accumulation chains over the 8 k-blocks.
    v8f t0 = {}, t1 = {};
#pragma unroll
    for (int kp = 0; kp < 4; ++kp) {
      t0 = __builtin_amdgcn_wmma_f32_16x16x32_bf16(
          false, afr[2 * kp + 0].v, false, bfr[2 * kp + 0].v,
          (short)0, t0, false, false);
      t1 = __builtin_amdgcn_wmma_f32_16x16x32_bf16(
          false, afr[2 * kp + 1].v, false, bfr[2 * kp + 1].v,
          (short)0, t1, false, false);
    }

    // Fold fHigh per output row (C layout: M = vgpr + hs*8).
#pragma unroll
    for (int r = 0; r < 8; ++r)
      acc[r] += fhBase[r * HPAD + h] * (t0[r] + t1[r]);
  }

  // ---- Phase 3: epilogue -------------------------------------------------
  const float bv = bias[o];
#pragma unroll
  for (int r = 0; r < 8; ++r) {
    const int row = g * MTILE + hs * 8 + r;
    out[(size_t)row * OUT_FEATURES + o] = acc[r] + bv;
  }
}

// ---------------------------------------------------------------------------
extern "C" void kernel_launch(void* const* d_in, const int* in_sizes, int n_in,
                              void* d_out, int out_size, void* d_ws, size_t ws_size,
                              hipStream_t stream) {
  (void)in_sizes; (void)n_in; (void)out_size;
  const float* x    = (const float*)d_in[0];   // [4096,16]
  const float* W    = (const float*)d_in[1];   // [64,65536]
  const float* bias = (const float*)d_in[2];   // [64]
  // d_in[3] = fbits (unused; basis enumerated analytically)
  float* out = (float*)d_out;

  const size_t WN   = (size_t)OUT_FEATURES * NBASIS;                       // 4,194,304
  const size_t smem = (size_t)MTILE * HPAD * 4 + (size_t)MTILE * LPAD * 2; // 25,088 B

  if (ws_size >= WN * sizeof(bf16_t)) {
    bf16_t* Wb = (bf16_t*)d_ws;
    wcvt_kernel<<<dim3((unsigned)(WN / 1024)), dim3(256), 0, stream>>>(W, Wb, (int)WN);
    lattice_gemm<true><<<dim3(BATCH / MTILE), dim3(128), smem, stream>>>(x, Wb, bias, out);
  } else {
    lattice_gemm<false><<<dim3(BATCH / MTILE), dim3(128), smem, stream>>>(x, W, bias, out);
  }
}
